// Decoder_10222022164898
// MI455X (gfx1250) — matrix operand order repaired and checked by reading, not compiled, not run on hardware
//
#include <hip/hip_runtime.h>
#include <math.h>

typedef __attribute__((ext_vector_type(16))) __bf16 v16bf;
typedef __attribute__((ext_vector_type(8)))  float  v8f;

// Problem constants (match reference)
#define BATCH 32
#define SEQ   2048
#define HID   1024
#define KDIM  3072            // 3*H (contraction dim)
#define MROWS (BATCH * SEQ)   // 65536

// ---------------------------------------------------------------------------
// K0: zero the normsq accumulator region
// ---------------------------------------------------------------------------
__global__ void zero_kernel(float* p, int n) {
    int i = blockIdx.x * blockDim.x + threadIdx.x;
    if (i < n) p[i] = 0.0f;
}

// ---------------------------------------------------------------------------
// K1: convert fc_w (f32 [H, K]) -> bf16 hi/lo pair (split precision)
// ---------------------------------------------------------------------------
__global__ void cvt_w_kernel(const float* __restrict__ fc_w,
                             unsigned short* __restrict__ whi,
                             unsigned short* __restrict__ wlo) {
    int i = blockIdx.x * blockDim.x + threadIdx.x;   // < H*K = 3,145,728
    float x  = fc_w[i];
    __bf16 h = (__bf16)x;
    __bf16 l = (__bf16)(x - (float)h);
    whi[i] = __builtin_bit_cast(unsigned short, h);
    wlo[i] = __builtin_bit_cast(unsigned short, l);
}

// ---------------------------------------------------------------------------
// K2: main GEMM (rows of t) via bf16 split WMMA; accumulate sum of squares
//     per row into normsq (t itself is never stored).
//     grid = (N/64, M/64), block = 128 (4 waves). Wave w -> rows m0+16w,
//     16x64 output tile per wave (4 WMMA n-tiles).
//     Inner loop is product-major: dependent WMMAs on the same accumulator
//     are separated by 3 independent WMMAs -> no hazard v_nop bubbles.
// ---------------------------------------------------------------------------
__global__ void __launch_bounds__(128)
gemm_norm_kernel(const float* __restrict__ trg,
                 const unsigned short* __restrict__ whi,
                 const unsigned short* __restrict__ wlo,
                 const float* __restrict__ fc_b,
                 float* __restrict__ normsq) {
    const int lane = threadIdx.x & 31;
    const int wave = threadIdx.x >> 5;          // 0..3
    const int half = lane >> 4;                 // 0 or 1
    const int l15  = lane & 15;                 // m-lane for A, n-lane for B/C

    const int m0 = blockIdx.y * 64 + wave * 16; // this wave's 16 rows
    const int n0 = blockIdx.x * 64;             // 4 n-tiles of 16

    // A: per-lane row pointer (row = m0 + l15), K-major contiguous
    const float* Arow = trg + (size_t)(m0 + l15) * KDIM;

    // B: per-lane row pointers for the 4 n-tiles (row = n, K-major contiguous)
    const unsigned short* bh_base[4];
    const unsigned short* bl_base[4];
#pragma unroll
    for (int t = 0; t < 4; ++t) {
        size_t off = (size_t)(n0 + 16 * t + l15) * KDIM + 16 * half;
        bh_base[t] = whi + off;
        bl_base[t] = wlo + off;
    }

    v8f acc[4] = {v8f{0}, v8f{0}, v8f{0}, v8f{0}};

    for (int k = 0; k < KDIM; k += 32) {
        // ---- load A fragment (fp32) in WMMA 16x32 bf16 layout ----
        const int koff = k + 16 * half;
        float4 a0 = *(const float4*)(Arow + koff);
        float4 a1 = *(const float4*)(Arow + koff + 4);
        float4 a2 = *(const float4*)(Arow + koff + 8);
        float4 a3 = *(const float4*)(Arow + koff + 12);
        float av[16] = {a0.x, a0.y, a0.z, a0.w, a1.x, a1.y, a1.z, a1.w,
                        a2.x, a2.y, a2.z, a2.w, a3.x, a3.y, a3.z, a3.w};
        v16bf ah{}, al{};
#pragma unroll
        for (int e = 0; e < 16; ++e) {
            __bf16 h = (__bf16)av[e];
            ah[e] = h;
            al[e] = (__bf16)(av[e] - (float)h);
        }
        // ---- load all 8 B fragments up front ----
        v16bf bh[4], bl[4];
#pragma unroll
        for (int t = 0; t < 4; ++t) {
            bh[t] = *(const v16bf*)(bh_base[t] + k);
            bl[t] = *(const v16bf*)(bl_base[t] + k);
        }
        // ---- product-major issue order: consecutive WMMAs independent ----
#pragma unroll
        for (int t = 0; t < 4; ++t)
            acc[t] = __builtin_amdgcn_wmma_f32_16x16x32_bf16(
                false, ah, false, bh[t], (short)0, acc[t], false, false);
#pragma unroll
        for (int t = 0; t < 4; ++t)
            acc[t] = __builtin_amdgcn_wmma_f32_16x16x32_bf16(
                false, ah, false, bl[t], (short)0, acc[t], false, false);
#pragma unroll
        for (int t = 0; t < 4; ++t)
            acc[t] = __builtin_amdgcn_wmma_f32_16x16x32_bf16(
                false, al, false, bh[t], (short)0, acc[t], false, false);
    }

    // ---- epilogue: add bias, square, reduce across the 16 lanes of each half
    // C layout: lane holds fixed n = l15; VGPR v holds row m = 8*half + v.
    float sq[8] = {0, 0, 0, 0, 0, 0, 0, 0};
#pragma unroll
    for (int t = 0; t < 4; ++t) {
        float bias = fc_b[n0 + 16 * t + l15];
#pragma unroll
        for (int v = 0; v < 8; ++v) {
            float x = acc[t][v] + bias;
            sq[v] += x * x;
        }
    }
#pragma unroll
    for (int v = 0; v < 8; ++v) {
        float r = sq[v];
        r += __shfl_xor(r, 1, 32);
        r += __shfl_xor(r, 2, 32);
        r += __shfl_xor(r, 4, 32);
        r += __shfl_xor(r, 8, 32);
        if (l15 == 0)
            atomicAdd(&normsq[m0 + 8 * half + v], r);
    }
}

// ---------------------------------------------------------------------------
// K3: per-batch: norm1 = sqrt(normsq) -> d_out; softmax over L -> wsoft
//     grid = 32, block = 256 (each thread handles 8 of the 2048 positions)
// ---------------------------------------------------------------------------
__global__ void softmax_kernel(const float* __restrict__ normsq,
                               float* __restrict__ norm1,
                               float* __restrict__ wsoft) {
    const int b = blockIdx.x, tid = threadIdx.x;
    __shared__ float red[256];
    const float* nsb = normsq + b * SEQ;

    float v[8];
    float mx = -1e30f;
#pragma unroll
    for (int i = 0; i < 8; ++i) {
        int l = i * 256 + tid;
        float nv = sqrtf(nsb[l]);
        v[i] = nv;
        norm1[b * SEQ + l] = nv;
        mx = fmaxf(mx, nv);
    }
    red[tid] = mx;
    __syncthreads();
    for (int s = 128; s > 0; s >>= 1) {
        if (tid < s) red[tid] = fmaxf(red[tid], red[tid + s]);
        __syncthreads();
    }
    mx = red[0];
    __syncthreads();

    float sum = 0.0f;
#pragma unroll
    for (int i = 0; i < 8; ++i) {
        v[i] = expf(v[i] - mx);
        sum += v[i];
    }
    red[tid] = sum;
    __syncthreads();
    for (int s = 128; s > 0; s >>= 1) {
        if (tid < s) red[tid] += red[tid + s];
        __syncthreads();
    }
    float inv = 1.0f / red[0];
#pragma unroll
    for (int i = 0; i < 8; ++i)
        wsoft[b * SEQ + i * 256 + tid] = v[i] * inv;
}

// ---------------------------------------------------------------------------
// K4: s[b,k] = sum_l wsoft[b,l] * trg[b,l,k]   (grid = (12, 32), block = 256)
// ---------------------------------------------------------------------------
__global__ void sweight_kernel(const float* __restrict__ trg,
                               const float* __restrict__ wsoft,
                               float* __restrict__ s) {
    const int b = blockIdx.y;
    const int k = blockIdx.x * 256 + threadIdx.x;
    __shared__ float wsh[SEQ];
    for (int i = threadIdx.x; i < SEQ; i += 256) wsh[i] = wsoft[b * SEQ + i];
    __syncthreads();

    const float* base = trg + (size_t)b * SEQ * KDIM + k;
    float acc = 0.0f;
#pragma unroll 4
    for (int l = 0; l < SEQ; ++l)
        acc = fmaf(wsh[l], base[(size_t)l * KDIM], acc);
    s[b * KDIM + k] = acc;
}

// ---------------------------------------------------------------------------
// K5: summ[b,h] = dot(s[b,:], fc_w[h,:]) + fc_b[h]  (grid = (4, 32), block=256)
// ---------------------------------------------------------------------------
__global__ void final_kernel(const float* __restrict__ s,
                             const float* __restrict__ fc_w,
                             const float* __restrict__ fc_b,
                             float* __restrict__ summ) {
    const int b = blockIdx.y;
    const int h = blockIdx.x * 256 + threadIdx.x;
    __shared__ float sh[KDIM];
    for (int i = threadIdx.x; i < KDIM; i += 256) sh[i] = s[b * KDIM + i];
    __syncthreads();

    const float* wr = fc_w + (size_t)h * KDIM;
    float acc = fc_b[h];
#pragma unroll 4
    for (int k = 0; k < KDIM; ++k)
        acc = fmaf(sh[k], wr[k], acc);
    summ[b * HID + h] = acc;
}

// ---------------------------------------------------------------------------
extern "C" void kernel_launch(void* const* d_in, const int* in_sizes, int n_in,
                              void* d_out, int out_size, void* d_ws, size_t ws_size,
                              hipStream_t stream) {
    const float* trg  = (const float*)d_in[0];
    // d_in[1] = src (unused, n_layers == 0)
    const float* fc_w = (const float*)d_in[2];
    const float* fc_b = (const float*)d_in[3];

    float* summ  = (float*)d_out;                 // [32, 1024]
    float* norm1 = (float*)d_out + BATCH * HID;   // [32, 2048]

    // workspace layout
    unsigned short* whi = (unsigned short*)d_ws;                 // bf16 hi [H*K]
    unsigned short* wlo = whi + (size_t)HID * KDIM;              // bf16 lo [H*K]
    float* normsq = (float*)(wlo + (size_t)HID * KDIM);          // [M]
    float* wsoft  = normsq + MROWS;                              // [B*L]
    float* s      = wsoft + MROWS;                               // [B*K]

    zero_kernel<<<MROWS / 256, 256, 0, stream>>>(normsq, MROWS);
    cvt_w_kernel<<<(HID * KDIM) / 256, 256, 0, stream>>>(fc_w, whi, wlo);
    gemm_norm_kernel<<<dim3(HID / 64, MROWS / 64), 128, 0, stream>>>(
        trg, whi, wlo, fc_b, normsq);
    softmax_kernel<<<BATCH, 256, 0, stream>>>(normsq, norm1, wsoft);
    sweight_kernel<<<dim3(KDIM / 256, BATCH), 256, 0, stream>>>(trg, wsoft, s);
    final_kernel<<<dim3(HID / 256, BATCH), 256, 0, stream>>>(s, fc_w, fc_b, summ);
}
